// GAT_11596411699260
// MI455X (gfx1250) — compile-verified, matrix operand order checked by reading, not run on hardware
//
#include <hip/hip_runtime.h>
#include <hip/hip_fp16.h>
#include <math.h>

typedef __attribute__((ext_vector_type(16))) _Float16 v16h;
typedef __attribute__((ext_vector_type(8)))  _Float16 v8h;
typedef __attribute__((ext_vector_type(8)))  float    v8f;

#define N_NODES   50000
#define N_EDGES   800000
#define E_TOT     (N_EDGES + N_NODES)
#define NEG_SLOPE 0.2f

// ---- float atomic max via ordered-int trick (works for +/- values) ----
__device__ __forceinline__ void atomicMaxF(float* addr, float v) {
    if (v >= 0.0f) atomicMax((int*)addr, __float_as_int(v));
    else           atomicMin((unsigned int*)addr, __float_as_uint(v));
}

__device__ __forceinline__ void edge_sd(int e, int* s, int* d, const int* __restrict__ ei) {
    if (e < N_EDGES) { *s = ei[e]; *d = ei[N_EDGES + e]; }
    else             { *s = *d = e - N_EDGES; }
}

// ---- layer 0 input: [N,6] f32 -> [N,32] f16 zero-padded ----
__global__ void k_x0_to_f16(const float* __restrict__ x, _Float16* __restrict__ xf) {
    int idx = blockIdx.x * blockDim.x + threadIdx.x;
    if (idx >= N_NODES * 32) return;
    int node = idx >> 5, j = idx & 31;
    float v = (j < 6) ? x[node * 6 + j] : 0.0f;
    xf[idx] = (_Float16)v;
}

// ---- W [K,Nout] f32 -> Wt [Npad,Kpad] f16 (transposed, zero-padded) ----
__global__ void k_w_to_f16t(const float* __restrict__ W, _Float16* __restrict__ Wt,
                            int K, int Nout, int Kpad, int Npad) {
    int idx = blockIdx.x * blockDim.x + threadIdx.x;
    if (idx >= Npad * Kpad) return;
    int nn = idx / Kpad, k = idx - nn * Kpad;
    float v = (nn < Nout && k < K) ? W[k * Nout + nn] : 0.0f;
    Wt[idx] = (_Float16)v;
}

// ---- WMMA GEMM: C[M,Npad] = A[M,K](f16) * Bt[Npad,K](f16)^T, f32 accum ----
// One wave per 16x16 output tile; per k-step each lane does two b128 loads
// per operand matching the CDNA5 16-bit A-matrix 16x32 layout.
__global__ void __launch_bounds__(128) k_gemm_wmma(
        const _Float16* __restrict__ A, const _Float16* __restrict__ Bt,
        float* __restrict__ Cm, int M, int K, int Npad) {
    int l    = threadIdx.x;
    int lane = l & 15;
    int lsel = l >> 4;
    int tn   = blockIdx.x;
    int tm   = blockIdx.y * blockDim.y + threadIdx.y;
    if (tm * 16 >= M) return;              // wave-uniform: EXEC stays all-1 for WMMA

    const _Float16* Ap = A  + (size_t)(tm * 16 + lane) * K + 8 * lsel;
    const _Float16* Bp = Bt + (size_t)(tn * 16 + lane) * K + 8 * lsel;

    v8f c = {};
    for (int k0 = 0; k0 < K; k0 += 32) {
        v8h alo = *(const v8h*)(Ap + k0);
        v8h ahi = *(const v8h*)(Ap + k0 + 16);
        v8h blo = *(const v8h*)(Bp + k0);
        v8h bhi = *(const v8h*)(Bp + k0 + 16);
        v16h a = __builtin_shufflevector(alo, ahi, 0,1,2,3,4,5,6,7,8,9,10,11,12,13,14,15);
        v16h b = __builtin_shufflevector(blo, bhi, 0,1,2,3,4,5,6,7,8,9,10,11,12,13,14,15);
        c = __builtin_amdgcn_wmma_f32_16x16x32_f16(false, a, false, b, (short)0, c, false, false);
    }
    // C/D layout: VGPR i -> row i (lanes 0-15) / i+8 (lanes 16-31), col = lane
    float* Cp = Cm + (size_t)(tm * 16 + 8 * lsel) * Npad + tn * 16 + lane;
#pragma unroll
    for (int i = 0; i < 8; ++i) Cp[(size_t)i * Npad] = c[i];
}

// ---- attention scalars per (node, head) + init emax/denom/acc ----
__global__ void k_alpha_init(const float* __restrict__ h, const float* __restrict__ a_src,
                             const float* __restrict__ a_dst, float* __restrict__ asrc,
                             float* __restrict__ adst, float* __restrict__ emax,
                             float* __restrict__ den, float* __restrict__ acc,
                             int H, int C, int ldh) {
    int idx = blockIdx.x * blockDim.x + threadIdx.x;
    if (idx >= N_NODES * H) return;
    int node = idx / H, hh = idx - node * H;
    const float* hp = h + (size_t)node * ldh + hh * C;
    float* ap = acc + (size_t)node * (H * C) + hh * C;
    float s1 = 0.f, s2 = 0.f;
    for (int cc = 0; cc < C; ++cc) {
        float v = hp[cc];
        s1 += v * a_src[hh * C + cc];
        s2 += v * a_dst[hh * C + cc];
        ap[cc] = 0.f;
    }
    asrc[idx] = s1;
    adst[idx] = s2;
    emax[idx] = -1e30f;
    den[idx]  = 0.f;
}

// ---- segment softmax pass 1: per-dst max of leaky_relu(e) ----
__global__ void k_edge_max(const int* __restrict__ ei, const float* __restrict__ asrc,
                           const float* __restrict__ adst, float* __restrict__ emax, int H) {
    int idx = blockIdx.x * blockDim.x + threadIdx.x;
    if (idx >= E_TOT * H) return;
    int e = idx / H, hh = idx - e * H;
    int s, d; edge_sd(e, &s, &d, ei);
    float v = asrc[s * H + hh] + adst[d * H + hh];
    v = (v >= 0.f) ? v : NEG_SLOPE * v;
    atomicMaxF(&emax[d * H + hh], v);
}

// ---- segment softmax pass 2: per-dst sum of exp(e - emax) ----
__global__ void k_edge_sum(const int* __restrict__ ei, const float* __restrict__ asrc,
                           const float* __restrict__ adst, const float* __restrict__ emax,
                           float* __restrict__ den, int H) {
    int idx = blockIdx.x * blockDim.x + threadIdx.x;
    if (idx >= E_TOT * H) return;
    int e = idx / H, hh = idx - e * H;
    int s, d; edge_sd(e, &s, &d, ei);
    float v = asrc[s * H + hh] + adst[d * H + hh];
    v = (v >= 0.f) ? v : NEG_SLOPE * v;
    atomicAdd(&den[d * H + hh], __expf(v - emax[d * H + hh]));
}

// ---- pass 3: weighted message scatter acc[dst] += alpha * h[src] ----
__global__ void k_edge_msg(const int* __restrict__ ei, const float* __restrict__ asrc,
                           const float* __restrict__ adst, const float* __restrict__ emax,
                           const float* __restrict__ den, const float* __restrict__ h,
                           float* __restrict__ acc, int H, int C, int ldh) {
    int HC  = H * C;
    int idx = blockIdx.x * blockDim.x + threadIdx.x;
    if (idx >= E_TOT * HC) return;
    int e = idx / HC, f = idx - e * HC;
    int hh = f / C;
    int s, d; edge_sd(e, &s, &d, ei);
    if (f == 0) __builtin_prefetch(&h[(size_t)s * ldh], 0, 0);   // global_prefetch_b8
    float v = asrc[s * H + hh] + adst[d * H + hh];
    v = (v >= 0.f) ? v : NEG_SLOPE * v;
    float alpha = __expf(v - emax[d * H + hh]) / den[d * H + hh];
    atomicAdd(&acc[(size_t)d * HC + f], h[(size_t)s * ldh + f] * alpha);
}

// ---- bias + (ELU -> f16 next-layer input) or final f32 output ----
__global__ void k_finish(const float* __restrict__ acc, const float* __restrict__ bias,
                         _Float16* __restrict__ xf_next, float* __restrict__ outp,
                         int HC, int last) {
    int idx = blockIdx.x * blockDim.x + threadIdx.x;
    if (idx >= N_NODES * HC) return;
    int node = idx / HC, f = idx - node * HC;
    float y = acc[idx] + bias[f];
    if (last) {
        outp[idx] = y;                               // HC == 6, matches d_out layout
    } else {
        y = (y > 0.f) ? y : expm1f(y);               // ELU
        xf_next[(size_t)node * 256 + f] = (_Float16)y;
    }
}

extern "C" void kernel_launch(void* const* d_in, const int* in_sizes, int n_in,
                              void* d_out, int out_size, void* d_ws, size_t ws_size,
                              hipStream_t stream) {
    (void)in_sizes; (void)n_in; (void)out_size; (void)ws_size;
    const float* x  = (const float*)d_in[0];
    const int*   ei = (const int*)d_in[1];

    char* ws = (char*)d_ws;
    size_t off = 0;
    auto carve = [&](size_t bytes) -> void* {
        void* p = ws + off;
        off = (off + bytes + 255) & ~(size_t)255;
        return p;
    };
    _Float16* xf   = (_Float16*)carve((size_t)N_NODES * 256 * sizeof(_Float16));
    float*    h    = (float*)   carve((size_t)N_NODES * 256 * sizeof(float));
    float*    acc  = (float*)   carve((size_t)N_NODES * 256 * sizeof(float));
    float*    asrc = (float*)   carve((size_t)N_NODES * 4   * sizeof(float));
    float*    adst = (float*)   carve((size_t)N_NODES * 4   * sizeof(float));
    float*    emax = (float*)   carve((size_t)N_NODES * 4   * sizeof(float));
    float*    den  = (float*)   carve((size_t)N_NODES * 4   * sizeof(float));
    _Float16* wt   = (_Float16*)carve((size_t)256 * 256 * sizeof(_Float16));

    const int Hs[4]   = {4, 4, 4, 1};
    const int Cs[4]   = {64, 64, 64, 6};
    const int Kin[4]  = {6, 256, 256, 256};
    const int Kpad[4] = {32, 256, 256, 256};

    // layer 0 input conversion (pad K: 6 -> 32)
    {
        int tot = N_NODES * 32;
        k_x0_to_f16<<<(tot + 255) / 256, 256, 0, stream>>>(x, xf);
    }

    for (int L = 0; L < 4; ++L) {
        const float* W    = (const float*)d_in[2 + 4 * L + 0];
        const float* a_s  = (const float*)d_in[2 + 4 * L + 1];
        const float* a_d  = (const float*)d_in[2 + 4 * L + 2];
        const float* bias = (const float*)d_in[2 + 4 * L + 3];
        int H = Hs[L], C = Cs[L], HC = H * C;
        int K = Kin[L], Kp = Kpad[L];
        int Np = (HC + 15) & ~15;

        {   // weight convert + transpose + pad
            int tot = Np * Kp;
            k_w_to_f16t<<<(tot + 255) / 256, 256, 0, stream>>>(W, wt, K, HC, Kp, Np);
        }
        {   // h = x @ W via WMMA (one wave per 16x16 tile, 4 waves/block)
            dim3 blk(32, 4);
            dim3 grd(Np / 16, (N_NODES / 16 + 3) / 4);
            k_gemm_wmma<<<grd, blk, 0, stream>>>(xf, wt, h, N_NODES, Kp, Np);
        }
        {   // attention scalars + state init
            int tot = N_NODES * H;
            k_alpha_init<<<(tot + 255) / 256, 256, 0, stream>>>(h, a_s, a_d, asrc, adst,
                                                                emax, den, acc, H, C, Np);
        }
        {   // segment softmax
            int tot = E_TOT * H;
            k_edge_max<<<(tot + 255) / 256, 256, 0, stream>>>(ei, asrc, adst, emax, H);
            k_edge_sum<<<(tot + 255) / 256, 256, 0, stream>>>(ei, asrc, adst, emax, den, H);
        }
        {   // message scatter
            long long tot = (long long)E_TOT * HC;
            unsigned nb = (unsigned)((tot + 255) / 256);
            k_edge_msg<<<nb, 256, 0, stream>>>(ei, asrc, adst, emax, den, h, acc, H, C, Np);
        }
        {   // bias (+ELU+f16) or final output
            int tot = N_NODES * HC;
            k_finish<<<(tot + 255) / 256, 256, 0, stream>>>(acc, bias, xf, (float*)d_out,
                                                            HC, (L == 3) ? 1 : 0);
        }
    }
}
